// NeuroGraph_74363063762963
// MI455X (gfx1250) — matrix operand order repaired and checked by reading, not compiled, not run on hardware
//
#include <hip/hip_runtime.h>

typedef __attribute__((ext_vector_type(16))) _Float16 v16h;
typedef __attribute__((ext_vector_type(8)))  float    v8f;

#define NN 100000
#define NE 400000
#define HS 100
#define TILES (NN / 16)   // 6250, exact
#define SV 512

union Frag { v16h v; _Float16 h[16]; };

__device__ __forceinline__ v8f wmma_f16(v16h a, v16h b, v8f c) {
  // D = A(16x32 f16) * B(32x16 f16) + C(16x16 f32)
  return __builtin_amdgcn_wmma_f32_16x16x32_f16(false, a, false, b, (short)0, c, false, false);
}

// ---- prepack weight W (K x P, row-major f32) into WMMA B-fragments (f16) ----
// B layout per 32-K block: h=0..7 -> k = kb + 8*(lane>>4) + h, h=8..15 -> k = kb + 16 + 8*(lane>>4) + (h-8);
// column n = ntile*16 + (lane&15).
__global__ void pack_w_kernel(const float* __restrict__ W, _Float16* __restrict__ dst,
                              int K, int P) {
  int frag = blockIdx.x;            // ntiles*4 fragments
  int nt = frag >> 2, kt = frag & 3;
  int lane = threadIdx.x;
  int p = nt * 16 + (lane & 15);
  int kb = kt * 32 + ((lane >> 4) << 3);
  Frag f;
#pragma unroll
  for (int h = 0; h < 16; ++h) {
    int k = kb + h + (h >= 8 ? 8 : 0);
    f.h[h] = (_Float16)((k < K && p < P) ? W[(size_t)k * P + p] : 0.f);
  }
  *(v16h*)(dst + ((size_t)frag * 32 + lane) * 16) = f.v;
}

// ---- vs0 = x @ type_vec (x is one-hot over 7 types) ----
__global__ void init_vs_kernel(const float* __restrict__ x, const float* __restrict__ tv,
                               float* __restrict__ vs) {
  int idx = blockIdx.x * blockDim.x + threadIdx.x;
  if (idx >= NN * HS) return;
  int i = idx / HS, j = idx - i * HS;
  float s = 0.f;
#pragma unroll
  for (int t = 0; t < 7; ++t) s += x[i * 7 + t] * tv[t * HS + j];
  vs[idx] = s;
}

__global__ void inv_deg_kernel(const float* a, const float* b, float* ia, float* ib) {
  int i = blockIdx.x * blockDim.x + threadIdx.x;
  if (i >= NN) return;
  ia[i] = 1.f / a[i];
  ib[i] = 1.f / b[i];
}

// ---- pack vs (f32 row-major) into A-fragment layout (f16, K padded 100->128) ----
// A layout: m = lane&15; h=0..7 -> k = kb + h, h=8..15 -> k = kb + 16 + (h-8),
// kb = kt*32 + 8*(lane>>4).
__global__ void pack_vs_kernel(const float* __restrict__ vs, _Float16* __restrict__ vsp) {
  int tile = blockIdx.x;
  int lane = threadIdx.x;
  int m = lane & 15;
  const float* rp = vs + ((size_t)tile * 16 + m) * HS;
#pragma unroll
  for (int kt = 0; kt < 4; ++kt) {
    int kb = kt * 32 + ((lane >> 4) << 3);
    Frag f;
#pragma unroll
    for (int j = 0; j < 8; ++j) {
      int k1 = kb + j, k2 = kb + 16 + j;
      f.h[j]     = (_Float16)((k1 < HS) ? rp[k1] : 0.f);
      f.h[8 + j] = (_Float16)((k2 < HS) ? rp[k2] : 0.f);
    }
    *(v16h*)(vsp + (((size_t)tile * 4 + kt) * 32 + lane) * 16) = f.v;
  }
}

// ---- fused 3-layer MLP over one 16-row tile per wave ----
__global__ void mlp_kernel(const _Float16* __restrict__ vsp,
                           const _Float16* __restrict__ w1p, const float* __restrict__ b1,
                           const _Float16* __restrict__ w2p, const float* __restrict__ b2,
                           const _Float16* __restrict__ w3p, const float* __restrict__ b3,
                           float* __restrict__ mout) {
  __shared__ _Float16 stage[16 * 136];   // 16 rows x 128 K (stride 136 halfs, conflict-free)
  int lane = threadIdx.x;
  int tile = blockIdx.x;
  for (int idx = lane; idx < 16 * 136; idx += 32) stage[idx] = (_Float16)0.f;
  __syncthreads();

  v16h a[4];
#pragma unroll
  for (int kt = 0; kt < 4; ++kt)
    a[kt] = *(const v16h*)(vsp + (((size_t)tile * 4 + kt) * 32 + lane) * 16);

  int colc  = lane & 15;
  int rhalf = (lane >> 4) << 3;  // C/D: VGPR r holds row r + 8*(lane>=16)

  const _Float16* wl[2] = {w1p, w2p};
  const float*    bl[2] = {b1, b2};
  for (int layer = 0; layer < 2; ++layer) {
    const _Float16* wp = wl[layer];
    const float*    bb = bl[layer];
    for (int nt = 0; nt < 7; ++nt) {
      v8f acc = {};
#pragma unroll
      for (int kt = 0; kt < 4; ++kt) {
        v16h bf = *(const v16h*)(wp + (((size_t)nt * 4 + kt) * 32 + lane) * 16);
        acc = wmma_f16(a[kt], bf, acc);
      }
      int col = nt * 16 + colc;
      float bias = (col < HS) ? bb[col] : 0.f;
#pragma unroll
      for (int r = 0; r < 8; ++r) {
        float v = acc[r] + bias;
        stage[(r + rhalf) * 136 + col] = (_Float16)(v > 0.f ? v : 0.f);  // relu
      }
    }
    __syncthreads();
    int m = lane & 15;
#pragma unroll
    for (int kt = 0; kt < 4; ++kt) {
      int kb = kt * 32 + ((lane >> 4) << 3);
      Frag f;
#pragma unroll
      for (int j = 0; j < 8; ++j) {
        f.h[j]     = stage[m * 136 + kb + j];
        f.h[8 + j] = stage[m * 136 + kb + 16 + j];
      }
      a[kt] = f.v;
    }
    __syncthreads();
  }
  // layer 3, no relu, straight to global (row-major f32 for the edge gather)
  for (int nt = 0; nt < 7; ++nt) {
    v8f acc = {};
#pragma unroll
    for (int kt = 0; kt < 4; ++kt) {
      v16h bf = *(const v16h*)(w3p + (((size_t)nt * 4 + kt) * 32 + lane) * 16);
      acc = wmma_f16(a[kt], bf, acc);
    }
    int col = nt * 16 + colc;
    if (col < HS) {
      float bias = b3[col];
#pragma unroll
      for (int r = 0; r < 8; ++r) {
        size_t row = (size_t)tile * 16 + r + rhalf;
        mout[row * HS + col] = acc[r] + bias;
      }
    }
  }
}

__global__ void zero_kernel(float* p, int n) {
  int i = blockIdx.x * blockDim.x + threadIdx.x;
  if (i < n) p[i] = 0.f;
}

// ---- edge segment-sum: agg[sidx[e]] += msg[gidx[e]] (4 cols / thread, L2-resident) ----
__global__ void scatter_kernel(const float* __restrict__ msg, const int* __restrict__ gidx,
                               const int* __restrict__ sidx, float* __restrict__ agg) {
  int t = blockIdx.x * blockDim.x + threadIdx.x;
  if (t >= NE * 25) return;
  int e = t / 25, q = t - e * 25;
  int s = gidx[e], d = sidx[e];
  float4 v = *(const float4*)(msg + (size_t)s * HS + q * 4);
  float* dp = agg + (size_t)d * HS + q * 4;
  unsafeAtomicAdd(dp + 0, v.x);
  unsafeAtomicAdd(dp + 1, v.y);
  unsafeAtomicAdd(dp + 2, v.z);
  unsafeAtomicAdd(dp + 3, v.w);
}

// ---- fused GRU: two GEMMs (N x 100 @ 100 x 300) + gates, per 16-row tile per wave ----
__global__ void gru_kernel(const float* __restrict__ agg, const float* __restrict__ invd,
                           const _Float16* __restrict__ wihp, const _Float16* __restrict__ whhp,
                           const float* __restrict__ bih, const float* __restrict__ bhh,
                           float* __restrict__ vs, _Float16* __restrict__ vsp) {
  __shared__ float gi[16 * 304];
  __shared__ float gh[16 * 304];
  __shared__ float nh[16 * 104];
  int lane  = threadIdx.x;
  int tile  = blockIdx.x;
  int m     = lane & 15;
  int rhalf = (lane >> 4) << 3;
  size_t row = (size_t)tile * 16 + m;
  float scale = invd[row];          // xin = segment_sum / deg, folded into A build
  const float* ap = agg + row * HS;

  v16h ax[4], ah[4];
#pragma unroll
  for (int kt = 0; kt < 4; ++kt) {
    int kb = kt * 32 + ((lane >> 4) << 3);
    float4 p0 = *(const float4*)(ap + kb);
    float4 p1 = *(const float4*)(ap + kb + 4);
    float4 p2 = *(const float4*)(ap + kb + 16);
    float4 p3 = *(const float4*)(ap + kb + 20);
    float t0[8] = {p0.x, p0.y, p0.z, p0.w, p1.x, p1.y, p1.z, p1.w};
    float t1[8] = {p2.x, p2.y, p2.z, p2.w, p3.x, p3.y, p3.z, p3.w};
    Frag f;
#pragma unroll
    for (int j = 0; j < 8; ++j) {
      f.h[j]     = (_Float16)(((kb + j)      < HS) ? t0[j] * scale : 0.f);
      f.h[8 + j] = (_Float16)(((kb + 16 + j) < HS) ? t1[j] * scale : 0.f);
    }
    ax[kt] = f.v;
    ah[kt] = *(const v16h*)(vsp + (((size_t)tile * 4 + kt) * 32 + lane) * 16);
  }

  for (int nt = 0; nt < 19; ++nt) {        // 304 cols cover 300
    v8f ai = {}, az = {};
#pragma unroll
    for (int kt = 0; kt < 4; ++kt) {
      v16h bi = *(const v16h*)(wihp + (((size_t)nt * 4 + kt) * 32 + lane) * 16);
      ai = wmma_f16(ax[kt], bi, ai);
      v16h bh = *(const v16h*)(whhp + (((size_t)nt * 4 + kt) * 32 + lane) * 16);
      az = wmma_f16(ah[kt], bh, az);
    }
    int col = nt * 16 + (lane & 15);
    if (col < 300) {
#pragma unroll
      for (int r = 0; r < 8; ++r) {
        gi[(r + rhalf) * 304 + col] = ai[r];
        gh[(r + rhalf) * 304 + col] = az[r];
      }
    }
  }
  __syncthreads();

  for (int idx = lane; idx < 16 * HS; idx += 32) {
    int rr = idx / HS, col = idx - rr * HS;
    size_t grow = (size_t)tile * 16 + rr;
    float ir = gi[rr * 304 + col]       + bih[col];
    float iz = gi[rr * 304 + col + 100] + bih[col + 100];
    float ig = gi[rr * 304 + col + 200] + bih[col + 200];
    float hr = gh[rr * 304 + col]       + bhh[col];
    float hz = gh[rr * 304 + col + 100] + bhh[col + 100];
    float hg = gh[rr * 304 + col + 200] + bhh[col + 200];
    float r = 1.f / (1.f + __expf(-(ir + hr)));
    float z = 1.f / (1.f + __expf(-(iz + hz)));
    float g = tanhf(ig + r * hg);
    float hold = vs[grow * HS + col];
    float hn = (1.f - z) * g + z * hold;
    nh[rr * 104 + col] = hn;
    vs[grow * HS + col] = hn;
  }
  __syncthreads();

  // re-emit packed f16 A-fragments of the new hidden state
#pragma unroll
  for (int kt = 0; kt < 4; ++kt) {
    int kb = kt * 32 + ((lane >> 4) << 3);
    Frag f;
#pragma unroll
    for (int j = 0; j < 8; ++j) {
      int k1 = kb + j, k2 = kb + 16 + j;
      f.h[j]     = (_Float16)((k1 < HS) ? nh[m * 104 + k1] : 0.f);
      f.h[8 + j] = (_Float16)((k2 < HS) ? nh[m * 104 + k2] : 0.f);
    }
    *(v16h*)(vsp + (((size_t)tile * 4 + kt) * 32 + lane) * 16) = f.v;
  }
}

// ---- final predictor on 512 shortcut nodes (tiny; scalar) ----
__global__ void predict_kernel(const float* __restrict__ vs, const int* __restrict__ svn,
                               const float* __restrict__ pw1, const float* __restrict__ pb1,
                               const float* __restrict__ pw2, const float* __restrict__ pb2,
                               float* __restrict__ out) {
  int i = blockIdx.x * blockDim.x + threadIdx.x;
  if (i >= SV) return;
  const float* r = vs + (size_t)svn[i] * HS;
  float o = pb2[0];
  for (int j = 0; j < 30; ++j) {
    float a = pb1[j];
    for (int k = 0; k < HS; ++k) a += r[k] * pw1[k * 30 + j];
    o += (a > 0.f ? a : 0.f) * pw2[j];
  }
  out[i] = o;
}

extern "C" void kernel_launch(void* const* d_in, const int* in_sizes, int n_in,
                              void* d_out, int out_size, void* d_ws, size_t ws_size,
                              hipStream_t stream) {
  (void)in_sizes; (void)n_in; (void)out_size; (void)ws_size;
  const float* x    = (const float*)d_in[0];
  const int*   ei   = (const int*)d_in[1];
  const int*   src  = ei;
  const int*   dst  = ei + NE;
  const int*   svn  = (const int*)d_in[2];
  const float* ind  = (const float*)d_in[3];
  const float* outd = (const float*)d_in[4];
  const float* tvec = (const float*)d_in[7];
  const float* fw1 = (const float*)d_in[8];  const float* fb1 = (const float*)d_in[9];
  const float* fw2 = (const float*)d_in[10]; const float* fb2 = (const float*)d_in[11];
  const float* fw3 = (const float*)d_in[12]; const float* fb3 = (const float*)d_in[13];
  const float* bw1 = (const float*)d_in[14]; const float* bb1 = (const float*)d_in[15];
  const float* bw2 = (const float*)d_in[16]; const float* bb2 = (const float*)d_in[17];
  const float* bw3 = (const float*)d_in[18]; const float* bb3 = (const float*)d_in[19];
  const float* fgwih = (const float*)d_in[20]; const float* fgwhh = (const float*)d_in[21];
  const float* fgbih = (const float*)d_in[22]; const float* fgbhh = (const float*)d_in[23];
  const float* bgwih = (const float*)d_in[24]; const float* bgwhh = (const float*)d_in[25];
  const float* bgbih = (const float*)d_in[26]; const float* bgbhh = (const float*)d_in[27];
  const float* pw1 = (const float*)d_in[28]; const float* pb1 = (const float*)d_in[29];
  const float* pw2 = (const float*)d_in[30]; const float* pb2 = (const float*)d_in[31];

  char* ws = (char*)d_ws;
  size_t off = 0;
  auto take = [&](size_t bytes) { char* p = ws + off; off += (bytes + 255) & ~(size_t)255; return p; };
  float*    vs   = (float*)take((size_t)NN * HS * 4);
  float*    mbuf = (float*)take((size_t)NN * HS * 4);
  float*    agg  = (float*)take((size_t)NN * HS * 4 + 4096);  // + pad for K-pad overreads
  _Float16* vsp  = (_Float16*)take((size_t)TILES * 4 * 32 * 16 * 2);
  float*    invi = (float*)take((size_t)NN * 4);
  float*    invo = (float*)take((size_t)NN * 4);
  const size_t P100 = (size_t)7 * 4 * 32 * 16;   // halfs per packed 100x100
  const size_t P300 = (size_t)19 * 4 * 32 * 16;  // halfs per packed 100x300
  _Float16* fw1p = (_Float16*)take(P100 * 2);
  _Float16* fw2p = (_Float16*)take(P100 * 2);
  _Float16* fw3p = (_Float16*)take(P100 * 2);
  _Float16* bw1p = (_Float16*)take(P100 * 2);
  _Float16* bw2p = (_Float16*)take(P100 * 2);
  _Float16* bw3p = (_Float16*)take(P100 * 2);
  _Float16* fgwihp = (_Float16*)take(P300 * 2);
  _Float16* fgwhhp = (_Float16*)take(P300 * 2);
  _Float16* bgwihp = (_Float16*)take(P300 * 2);
  _Float16* bgwhhp = (_Float16*)take(P300 * 2);

  pack_w_kernel<<<7 * 4, 32, 0, stream>>>(fw1, fw1p, HS, HS);
  pack_w_kernel<<<7 * 4, 32, 0, stream>>>(fw2, fw2p, HS, HS);
  pack_w_kernel<<<7 * 4, 32, 0, stream>>>(fw3, fw3p, HS, HS);
  pack_w_kernel<<<7 * 4, 32, 0, stream>>>(bw1, bw1p, HS, HS);
  pack_w_kernel<<<7 * 4, 32, 0, stream>>>(bw2, bw2p, HS, HS);
  pack_w_kernel<<<7 * 4, 32, 0, stream>>>(bw3, bw3p, HS, HS);
  pack_w_kernel<<<19 * 4, 32, 0, stream>>>(fgwih, fgwihp, HS, 300);
  pack_w_kernel<<<19 * 4, 32, 0, stream>>>(fgwhh, fgwhhp, HS, 300);
  pack_w_kernel<<<19 * 4, 32, 0, stream>>>(bgwih, bgwihp, HS, 300);
  pack_w_kernel<<<19 * 4, 32, 0, stream>>>(bgwhh, bgwhhp, HS, 300);

  init_vs_kernel<<<(NN * HS + 255) / 256, 256, 0, stream>>>(x, tvec, vs);
  inv_deg_kernel<<<(NN + 255) / 256, 256, 0, stream>>>(ind, outd, invi, invo);
  pack_vs_kernel<<<TILES, 32, 0, stream>>>(vs, vsp);

  const int zgrid = (NN * HS + 255) / 256;
  const int sgrid = (NE * 25 + 255) / 256;
  for (int round = 0; round < 8; ++round) {
    mlp_kernel<<<TILES, 32, 0, stream>>>(vsp, fw1p, fb1, fw2p, fb2, fw3p, fb3, mbuf);
    zero_kernel<<<zgrid, 256, 0, stream>>>(agg, NN * HS);
    scatter_kernel<<<sgrid, 256, 0, stream>>>(mbuf, src, dst, agg);   // fwd
    gru_kernel<<<TILES, 32, 0, stream>>>(agg, invi, fgwihp, fgwhhp, fgbih, fgbhh, vs, vsp);
    mlp_kernel<<<TILES, 32, 0, stream>>>(vsp, bw1p, bb1, bw2p, bb2, bw3p, bb3, mbuf);
    zero_kernel<<<zgrid, 256, 0, stream>>>(agg, NN * HS);
    scatter_kernel<<<sgrid, 256, 0, stream>>>(mbuf, dst, src, agg);   // bwd
    gru_kernel<<<TILES, 32, 0, stream>>>(agg, invo, bgwihp, bgwhhp, bgbih, bgbhh, vs, vsp);
  }
  predict_kernel<<<(SV + 255) / 256, 256, 0, stream>>>(vs, svn, pw1, pb1, pw2, pb2, (float*)d_out);
}
